// GCNClassifier_78786880077898
// MI455X (gfx1250) — compile-verified, matrix operand order checked by reading, not run on hardware
//
#include <hip/hip_runtime.h>
#include <hip/hip_bf16.h>

#define GCN_N 100000
#define GCN_E 1600000
#define F_IN 128
#define F_HID 256
#define F_OUT 32

typedef __attribute__((ext_vector_type(16))) __bf16        v16bf;
typedef __attribute__((ext_vector_type(8)))  float         v8f;
typedef __attribute__((ext_vector_type(8)))  unsigned int  v8u;

__device__ __forceinline__ unsigned short f32_to_bf16(float f) {
  unsigned int u = __builtin_bit_cast(unsigned int, f);
  u += 0x7FFFu + ((u >> 16) & 1u);           // round-to-nearest-even
  return (unsigned short)(u >> 16);
}
__device__ __forceinline__ float bf16_to_f32(unsigned short h) {
  return __builtin_bit_cast(float, (unsigned int)h << 16);
}

// ---------------- prep kernels ----------------
__global__ void k_zero_f32(float* __restrict__ p, int n) {
  int i = blockIdx.x * blockDim.x + threadIdx.x;
  if (i < n) p[i] = 0.0f;
}

__global__ void k_degree(const int* __restrict__ dst, float* __restrict__ deg, int e) {
  int i = blockIdx.x * blockDim.x + threadIdx.x;
  if (i < e) atomicAdd(&deg[dst[i]], 1.0f);
}

__global__ void k_dinv(float* __restrict__ deg, int n) {
  int i = blockIdx.x * blockDim.x + threadIdx.x;
  if (i < n) deg[i] = rsqrtf(deg[i] + 1.0f);   // deg includes self-loop
}

__global__ void k_f32_to_bf16(const float* __restrict__ in, unsigned short* __restrict__ out, int n) {
  int i = blockIdx.x * blockDim.x + threadIdx.x;
  if (i < n) out[i] = f32_to_bf16(in[i]);
}

// WT[c*K + k] = bf16(W[k*NO + c])  -> B fragments become contiguous b128 loads
__global__ void k_transpose_w(const float* __restrict__ W, unsigned short* __restrict__ WT,
                              int K, int NO) {
  int i = blockIdx.x * blockDim.x + threadIdx.x;
  if (i >= K * NO) return;
  int k = i / NO, c = i - k * NO;
  WT[(size_t)c * K + k] = f32_to_bf16(W[i]);
}

// ---------------- WMMA GEMM: H[M x NO] = A[M x K] @ W, bf16 in / f32 acc / bf16 out ----------------
__global__ void __launch_bounds__(256)
k_wmma_gemm_bf16(const unsigned short* __restrict__ A,
                 const unsigned short* __restrict__ WT,   // [NO x K] transposed
                 unsigned short* __restrict__ H,
                 int rowTiles, int K, int NO) {
  const int gw   = (int)((blockIdx.x * blockDim.x + threadIdx.x) >> 5); // global wave id
  const int lane = threadIdx.x & 31;
  const int colTiles = NO >> 4;
  const int rowTile  = gw / colTiles;
  const int colTile  = gw - rowTile * colTiles;
  if (rowTile >= rowTiles) return;              // wave-uniform: EXEC all-1s for WMMA

  const int half = lane >> 4;                   // K-interleave half (ISA 16-bit A/B layout)
  const int rc   = lane & 15;                   // A: row-in-tile, B: col-in-tile
  const unsigned short* arow = A  + (size_t)(rowTile * 16 + rc) * K;
  const unsigned short* brow = WT + (size_t)(colTile * 16 + rc) * K;

  v8f acc = {};
  for (int k0 = 0; k0 < K; k0 += 32) {
    // lane holds K = {8h..8h+7} and {16+8h..16+8h+7}: two contiguous 16B loads each
    const uint4 a0 = *(const uint4*)(arow + k0 + 8 * half);
    const uint4 a1 = *(const uint4*)(arow + k0 + 16 + 8 * half);
    const uint4 b0 = *(const uint4*)(brow + k0 + 8 * half);
    const uint4 b1 = *(const uint4*)(brow + k0 + 16 + 8 * half);
    v8u au = {a0.x, a0.y, a0.z, a0.w, a1.x, a1.y, a1.z, a1.w};
    v8u bu = {b0.x, b0.y, b0.z, b0.w, b1.x, b1.y, b1.z, b1.w};
    acc = __builtin_amdgcn_wmma_f32_16x16x32_bf16(
        false, __builtin_bit_cast(v16bf, au),
        false, __builtin_bit_cast(v16bf, bu),
        (short)0, acc, false, false);
  }
  // C/D layout: lane -> N = lane&15, VGPR p -> M = p + 8*(lane>>4)
  unsigned short* hout = H + (size_t)(rowTile * 16 + 8 * half) * NO + colTile * 16 + rc;
#pragma unroll
  for (int p = 0; p < 8; ++p)
    hout[(size_t)p * NO] = f32_to_bf16(acc[p]);
}

// ---------------- aggregation ----------------
// AGG[i,f] = dinv[i]^2 * H[i,f]   (self-loop term; also initializes the accumulator)
__global__ void k_self_init(const unsigned short* __restrict__ Hb, const float* __restrict__ dinv,
                            float* __restrict__ AGG, int fshift, int n) {
  int i = blockIdx.x * blockDim.x + threadIdx.x;
  if (i >= n) return;
  float di = dinv[i >> fshift];
  AGG[i] = di * di * bf16_to_f32(Hb[i]);
}

// AGG[dst,f] += dinv[src]*dinv[dst] * H[src,f]; blockIdx.y selects the 8-feature chunk
__global__ void __launch_bounds__(256)
k_edge_scatter(const int* __restrict__ src, const int* __restrict__ dst,
               const float* __restrict__ dinv, const unsigned short* __restrict__ Hb,
               float* __restrict__ AGG, int F) {
  int e = blockIdx.x * blockDim.x + threadIdx.x;
  if (e >= GCN_E) return;
  int ch = blockIdx.y;                       // F/8 chunks
  int s = src[e], d = dst[e];
  float coef = dinv[s] * dinv[d];
  const uint4 hv = *(const uint4*)(Hb + (size_t)s * F + ch * 8);
  float* op = AGG + (size_t)d * F + ch * 8;
  unsigned int w[4] = {hv.x, hv.y, hv.z, hv.w};
#pragma unroll
  for (int j = 0; j < 4; ++j) {
    atomicAdd(op + 2 * j,     coef * bf16_to_f32((unsigned short)(w[j] & 0xFFFFu)));
    atomicAdd(op + 2 * j + 1, coef * bf16_to_f32((unsigned short)(w[j] >> 16)));
  }
}

// out_bf16 = relu(bn(AGG + bias))  -> becomes next layer's A matrix
__global__ void k_bias_bn_relu_bf16(const float* __restrict__ AGG,
                                    const float* __restrict__ b,  const float* __restrict__ g,
                                    const float* __restrict__ be, const float* __restrict__ m,
                                    const float* __restrict__ v,
                                    unsigned short* __restrict__ Aout, int fmask, int n) {
  int i = blockIdx.x * blockDim.x + threadIdx.x;
  if (i >= n) return;
  int f = i & fmask;
  float x = AGG[i] + b[f];
  float y = (x - m[f]) * rsqrtf(v[f] + 1e-5f) * g[f] + be[f];
  Aout[i] = f32_to_bf16(fmaxf(y, 0.0f));
}

__global__ void k_bias_sigmoid(const float* __restrict__ AGG, const float* __restrict__ b,
                               float* __restrict__ out, int fmask, int n) {
  int i = blockIdx.x * blockDim.x + threadIdx.x;
  if (i >= n) return;
  float x = AGG[i] + b[i & fmask];
  out[i] = 1.0f / (1.0f + __expf(-x));
}

// ---------------- launch ----------------
extern "C" void kernel_launch(void* const* d_in, const int* in_sizes, int n_in,
                              void* d_out, int out_size, void* d_ws, size_t ws_size,
                              hipStream_t stream) {
  const float* x  = (const float*)d_in[0];
  const int*   ei = (const int*)d_in[1];       // [2,E] int32 (JAX default demotes int64)
  const int*   src = ei;
  const int*   dst = ei + GCN_E;
  const float* W1 = (const float*)d_in[2];  const float* b1 = (const float*)d_in[3];
  const float* g1 = (const float*)d_in[4];  const float* be1= (const float*)d_in[5];
  const float* m1 = (const float*)d_in[6];  const float* v1 = (const float*)d_in[7];
  const float* W2 = (const float*)d_in[8];  const float* b2 = (const float*)d_in[9];
  const float* g2 = (const float*)d_in[10]; const float* be2= (const float*)d_in[11];
  const float* m2 = (const float*)d_in[12]; const float* v2 = (const float*)d_in[13];
  const float* W3 = (const float*)d_in[14]; const float* b3 = (const float*)d_in[15];
  float* out = (float*)d_out;

  // workspace layout (256B aligned)
  char* ws = (char*)d_ws;
  size_t off = 0;
  float* dinv = (float*)(ws + off);           off += ((size_t)GCN_N * 4 + 255) & ~(size_t)255;
  unsigned short* Abf  = (unsigned short*)(ws + off); off += (size_t)GCN_N * F_HID * 2;
  unsigned short* Hbf  = (unsigned short*)(ws + off); off += (size_t)GCN_N * F_HID * 2;
  unsigned short* WTbf = (unsigned short*)(ws + off); off += (size_t)F_HID * F_HID * 2;
  float* AGG = (float*)(ws + off);            off += (size_t)GCN_N * F_HID * 4;
  (void)off; (void)ws_size; (void)n_in; (void)in_sizes; (void)out_size;

  const int B = 256;
  const int rowTiles = GCN_N / 16;            // 6250, exact

  // degrees -> dinv (in place)
  k_zero_f32<<<(GCN_N + B - 1) / B, B, 0, stream>>>(dinv, GCN_N);
  k_degree<<<GCN_E / B, B, 0, stream>>>(dst, dinv, GCN_E);
  k_dinv<<<(GCN_N + B - 1) / B, B, 0, stream>>>(dinv, GCN_N);

  // A = bf16(x)
  k_f32_to_bf16<<<(GCN_N * F_IN) / B, B, 0, stream>>>(x, Abf, GCN_N * F_IN);

  // ---- layer 1: 128 -> 256 ----
  k_transpose_w<<<(F_IN * F_HID) / B, B, 0, stream>>>(W1, WTbf, F_IN, F_HID);
  k_wmma_gemm_bf16<<<(rowTiles * (F_HID / 16) + 7) / 8, B, 0, stream>>>(Abf, WTbf, Hbf, rowTiles, F_IN, F_HID);
  k_self_init<<<(GCN_N * F_HID) / B, B, 0, stream>>>(Hbf, dinv, AGG, 8, GCN_N * F_HID);
  k_edge_scatter<<<dim3(GCN_E / B, F_HID / 8), B, 0, stream>>>(src, dst, dinv, Hbf, AGG, F_HID);
  k_bias_bn_relu_bf16<<<(GCN_N * F_HID) / B, B, 0, stream>>>(AGG, b1, g1, be1, m1, v1, Abf, F_HID - 1, GCN_N * F_HID);

  // ---- layer 2: 256 -> 256 ----
  k_transpose_w<<<(F_HID * F_HID) / B, B, 0, stream>>>(W2, WTbf, F_HID, F_HID);
  k_wmma_gemm_bf16<<<(rowTiles * (F_HID / 16) + 7) / 8, B, 0, stream>>>(Abf, WTbf, Hbf, rowTiles, F_HID, F_HID);
  k_self_init<<<(GCN_N * F_HID) / B, B, 0, stream>>>(Hbf, dinv, AGG, 8, GCN_N * F_HID);
  k_edge_scatter<<<dim3(GCN_E / B, F_HID / 8), B, 0, stream>>>(src, dst, dinv, Hbf, AGG, F_HID);
  k_bias_bn_relu_bf16<<<(GCN_N * F_HID) / B, B, 0, stream>>>(AGG, b2, g2, be2, m2, v2, Abf, F_HID - 1, GCN_N * F_HID);

  // ---- layer 3: 256 -> 32, sigmoid ----
  k_transpose_w<<<(F_HID * F_OUT) / B, B, 0, stream>>>(W3, WTbf, F_HID, F_OUT);
  k_wmma_gemm_bf16<<<(rowTiles * (F_OUT / 16) + 7) / 8, B, 0, stream>>>(Abf, WTbf, Hbf, rowTiles, F_HID, F_OUT);
  k_self_init<<<(GCN_N * F_OUT) / B, B, 0, stream>>>(Hbf, dinv, AGG, 5, GCN_N * F_OUT);
  k_edge_scatter<<<dim3(GCN_E / B, F_OUT / 8), B, 0, stream>>>(src, dst, dinv, Hbf, AGG, F_OUT);
  k_bias_sigmoid<<<(GCN_N * F_OUT) / B, B, 0, stream>>>(AGG, b3, out, F_OUT - 1, GCN_N * F_OUT);
}
